// StateModelEncoder_5798205850200
// MI455X (gfx1250) — matrix operand order repaired and checked by reading, not compiled, not run on hardware
//
#include <hip/hip_runtime.h>
#include <math.h>

typedef __attribute__((ext_vector_type(16))) _Float16 v16h;
typedef __attribute__((ext_vector_type(8)))  float    v8f;

#define NGAME 50000
#define NSTATE 20000
#define NPC 30000

// ---------------------------------------------------------------------------
// WMMA GEMM: C[N x COLS] = A[N x 128] @ W[128 x COLS] (+bias) (+Cin) (opt relu)
// Block = 256 threads (8 waves). Block tile = 128 rows x COLS cols.
// Wave w computes rows [16w, 16w+16); NCT column tiles of 16.
// ---------------------------------------------------------------------------
template<int NCT, bool RELU>
__global__ __launch_bounds__(256)
void gemm128_wmma(const float* __restrict__ A, const float* __restrict__ W,
                  const float* __restrict__ bias, const float* __restrict__ Cin,
                  float* __restrict__ C, int N) {
  constexpr int COLS = NCT * 16;
  __shared__ _Float16 Af[128][34];        // 128 rows x 32 k (padded)
  __shared__ _Float16 Bf[32][COLS + 8];   // 32 k x COLS (padded)
  const int t = threadIdx.x;
  const int wave = t >> 5;
  const int lane = t & 31;
  const int rowBase = blockIdx.x * 128;

  v8f acc[NCT];
#pragma unroll
  for (int i = 0; i < NCT; ++i)
#pragma unroll
    for (int j = 0; j < 8; ++j) acc[i][j] = 0.0f;

  for (int kc = 0; kc < 4; ++kc) {
    // stage A tile (128x32 fp32 -> f16)
    for (int i = t; i < 128 * 32; i += 256) {
      int r = i >> 5, k = i & 31;
      int gr = rowBase + r;
      float v = (gr < N) ? A[(size_t)gr * 128 + kc * 32 + k] : 0.0f;
      Af[r][k] = (_Float16)v;
    }
    // stage W tile (32 x COLS)
    for (int i = t; i < 32 * COLS; i += 256) {
      int k = i / COLS, c = i % COLS;
      Bf[k][c] = (_Float16)W[(size_t)(kc * 32 + k) * COLS + c];
    }
    __syncthreads();

    // A fragment: lane row = 16w + (lane&15); K per ISA 16-bit A 16x32 table
    v16h a;
    const int arow = wave * 16 + (lane & 15);
#pragma unroll
    for (int j = 0; j < 16; ++j) {
      int kk = ((j < 8) ? j : (j + 8)) + ((lane >= 16) ? 8 : 0);
      a[j] = Af[arow][kk];
    }
    // B fragment: lane holds column N = lane&15; K per same table order as A
    const int bk = (lane & 15);
#pragma unroll
    for (int tt = 0; tt < NCT; ++tt) {
      v16h b;
#pragma unroll
      for (int j = 0; j < 16; ++j) b[j] = Bf[((j < 8) ? j : (j + 8)) + ((lane >= 16) ? 8 : 0)][tt * 16 + bk];
      acc[tt] = __builtin_amdgcn_wmma_f32_16x16x32_f16(
          false, a, false, b, (short)0, acc[tt], false, false);
    }
    __syncthreads();
  }

  // epilogue: VGPR i -> row i (+8 if lane>=16), N = lane&15
#pragma unroll
  for (int tt = 0; tt < NCT; ++tt) {
    const int col = tt * 16 + (lane & 15);
#pragma unroll
    for (int i = 0; i < 8; ++i) {
      int r = rowBase + wave * 16 + i + ((lane >= 16) ? 8 : 0);
      if (r < N) {
        float v = acc[tt][i];
        if (bias) v += bias[col];
        if (Cin)  v += Cin[(size_t)r * COLS + col];
        if (RELU) v = fmaxf(v, 0.0f);
        C[(size_t)r * COLS + col] = v;
      }
    }
  }
}

// ---------------------------------------------------------------------------
// Small-K dense: out[n][j] (+)= sum_c x[n][c] * W[c][j] (+bias)   (K<=8)
// ---------------------------------------------------------------------------
__global__ void gemm_smallK(const float* __restrict__ x, const float* __restrict__ W,
                            const float* __restrict__ bias, float* __restrict__ out,
                            int N, int K, int accum) {
  long long total = (long long)N * 128;
  for (long long idx = blockIdx.x * (long long)blockDim.x + threadIdx.x; idx < total;
       idx += (long long)gridDim.x * blockDim.x) {
    int n = (int)(idx >> 7), j = (int)(idx & 127);
    float s = accum ? out[idx] : 0.0f;
    if (bias) s += bias[j];
    for (int c = 0; c < K; ++c) s += x[(size_t)n * K + c] * W[c * 128 + j];
    out[idx] = s;
  }
}

// ------------------------- elementwise / utility ---------------------------
__global__ void zero_k(float* p, long long n) {
  for (long long i = blockIdx.x * (long long)blockDim.x + threadIdx.x; i < n;
       i += (long long)gridDim.x * blockDim.x) p[i] = 0.0f;
}
__global__ void relu_k(float* p, long long n) {
  for (long long i = blockIdx.x * (long long)blockDim.x + threadIdx.x; i < n;
       i += (long long)gridDim.x * blockDim.x) p[i] = fmaxf(p[i], 0.0f);
}
__global__ void add_relu_k(float* a, const float* b, long long n) {
  for (long long i = blockIdx.x * (long long)blockDim.x + threadIdx.x; i < n;
       i += (long long)gridDim.x * blockDim.x) a[i] = fmaxf(a[i] + b[i], 0.0f);
}
__global__ void div_cnt_k(float* m, const float* cnt, long long n128) {
  for (long long i = blockIdx.x * (long long)blockDim.x + threadIdx.x; i < n128;
       i += (long long)gridDim.x * blockDim.x) m[i] /= fmaxf(cnt[i >> 7], 1.0f);
}
__global__ void gcn_epi_k(float* agg, const float* xW, const float* dinv,
                          const float* bias, long long n128) {
  for (long long i = blockIdx.x * (long long)blockDim.x + threadIdx.x; i < n128;
       i += (long long)gridDim.x * blockDim.x) {
    int n = (int)(i >> 7), c = (int)(i & 127);
    float di = dinv[n];
    agg[i] = fmaxf(agg[i] + di * di * xW[i] + bias[c], 0.0f);
  }
}
__global__ void dinv_k(float* deg, int n, float addSelf) {
  for (long long i = blockIdx.x * (long long)blockDim.x + threadIdx.x; i < n;
       i += (long long)gridDim.x * blockDim.x) {
    float d = deg[i] + addSelf;
    deg[i] = (d > 0.0f) ? rsqrtf(d) : 0.0f;
  }
}

// ------------------------------ edge kernels -------------------------------
__global__ void count_deg_k(const int* __restrict__ col, int E, float* deg) {
  for (long long e = blockIdx.x * (long long)blockDim.x + threadIdx.x; e < E;
       e += (long long)gridDim.x * blockDim.x) atomicAdd(&deg[col[e]], 1.0f);
}
__global__ void rgcn_count_k(const int* __restrict__ col, const int* __restrict__ et,
                             int E, float* cnt3) {
  for (long long e = blockIdx.x * (long long)blockDim.x + threadIdx.x; e < E;
       e += (long long)gridDim.x * blockDim.x)
    atomicAdd(&cnt3[(size_t)col[e] * 3 + et[e]], 1.0f);
}
__global__ void prop_norm_k(const float* __restrict__ h, const int* __restrict__ row,
                            const int* __restrict__ col, const float* __restrict__ dinv,
                            float* __restrict__ out, long long total, int C) {
  for (long long idx = blockIdx.x * (long long)blockDim.x + threadIdx.x; idx < total;
       idx += (long long)gridDim.x * blockDim.x) {
    int e = (int)(idx / C), c = (int)(idx % C);
    int r = row[e], d = col[e];
    atomicAdd(&out[(size_t)d * C + c], dinv[r] * dinv[d] * h[(size_t)r * C + c]);
  }
}
__global__ void sum_rows_k(const float* __restrict__ src, const int* __restrict__ row,
                           const int* __restrict__ col, float* __restrict__ out,
                           long long total) {  // C == 128
  for (long long idx = blockIdx.x * (long long)blockDim.x + threadIdx.x; idx < total;
       idx += (long long)gridDim.x * blockDim.x) {
    int e = (int)(idx >> 7), c = (int)(idx & 127);
    atomicAdd(&out[(size_t)col[e] * 128 + c], src[(size_t)row[e] * 128 + c]);
  }
}
__global__ void rgcn_scatter_k(const float* __restrict__ Hr, const int* __restrict__ row,
                               const int* __restrict__ col, const int* __restrict__ et,
                               const float* __restrict__ cnt3, int r, float* __restrict__ out,
                               long long total) {
  for (long long idx = blockIdx.x * (long long)blockDim.x + threadIdx.x; idx < total;
       idx += (long long)gridDim.x * blockDim.x) {
    int e = (int)(idx >> 7), c = (int)(idx & 127);
    if (et[e] != r) continue;
    int d = col[e];
    float w = 1.0f / fmaxf(cnt3[(size_t)d * 3 + r], 1.0f);
    atomicAdd(&out[(size_t)d * 128 + c], Hr[(size_t)row[e] * 128 + c] * w);
  }
}
// fused ResGated edge kernel: eta = sigmoid(kk[dst]+q[src]+edge@We); out += eta*v[src]
__global__ void gated_k(const float* __restrict__ kk, const float* __restrict__ q,
                        const float* __restrict__ v, const float* __restrict__ ea,
                        const float* __restrict__ we, const int* __restrict__ row,
                        const int* __restrict__ col, float* __restrict__ out,
                        long long total) {
  for (long long idx = blockIdx.x * (long long)blockDim.x + threadIdx.x; idx < total;
       idx += (long long)gridDim.x * blockDim.x) {
    int e = (int)(idx >> 7), c = (int)(idx & 127);
    int r = row[e], d = col[e];
    float x = kk[(size_t)d * 128 + c] + q[(size_t)r * 128 + c] +
              ea[(size_t)e * 2] * we[c] + ea[(size_t)e * 2 + 1] * we[128 + c];
    float eta = 1.0f / (1.0f + expf(-x));
    atomicAdd(&out[(size_t)d * 128 + c], eta * v[(size_t)r * 128 + c]);
  }
}

// per-row L2 normalize then relu (one 128-thread block per row)
__global__ void l2norm_relu_k(float* x, int N) {
  int row = blockIdx.x;
  if (row >= N) return;
  float v = x[(size_t)row * 128 + threadIdx.x];
  __shared__ float red[128];
  red[threadIdx.x] = v * v;
  __syncthreads();
  for (int s = 64; s > 0; s >>= 1) {
    if (threadIdx.x < s) red[threadIdx.x] += red[threadIdx.x + s];
    __syncthreads();
  }
  float nrm = fmaxf(sqrtf(red[0]), 1e-12f);
  x[(size_t)row * 128 + threadIdx.x] = fmaxf(v / nrm, 0.0f);
}

__global__ void rowdot32_k(const float* __restrict__ s, const float* __restrict__ w,
                           const float* __restrict__ b, float* __restrict__ logits, int N) {
  int n = blockIdx.x * blockDim.x + threadIdx.x;
  if (n >= N) return;
  float acc = b[0];
#pragma unroll
  for (int j = 0; j < 32; ++j) acc += s[(size_t)n * 32 + j] * w[j];
  logits[n] = acc;
}

// ------------------------- log_softmax reductions --------------------------
__global__ void redmax_k(const float* x, int n, float* part) {
  __shared__ float red[256];
  float m = -INFINITY;
  for (int i = blockIdx.x * 256 + threadIdx.x; i < n; i += gridDim.x * 256)
    m = fmaxf(m, x[i]);
  red[threadIdx.x] = m; __syncthreads();
  for (int s = 128; s > 0; s >>= 1) {
    if (threadIdx.x < s) red[threadIdx.x] = fmaxf(red[threadIdx.x], red[threadIdx.x + s]);
    __syncthreads();
  }
  if (threadIdx.x == 0) part[blockIdx.x] = red[0];
}
__global__ void finmax_k(float* part) {  // 64 threads, 1 block -> part[64]
  __shared__ float red[64];
  red[threadIdx.x] = part[threadIdx.x]; __syncthreads();
  for (int s = 32; s > 0; s >>= 1) {
    if (threadIdx.x < s) red[threadIdx.x] = fmaxf(red[threadIdx.x], red[threadIdx.x + s]);
    __syncthreads();
  }
  if (threadIdx.x == 0) part[64] = red[0];
}
__global__ void redsum_k(const float* x, int n, const float* mx, float* part) {
  __shared__ float red[256];
  float m = mx[0], s = 0.0f;
  for (int i = blockIdx.x * 256 + threadIdx.x; i < n; i += gridDim.x * 256)
    s += expf(x[i] - m);
  red[threadIdx.x] = s; __syncthreads();
  for (int t = 128; t > 0; t >>= 1) {
    if (threadIdx.x < t) red[threadIdx.x] += red[threadIdx.x + t];
    __syncthreads();
  }
  if (threadIdx.x == 0) part[blockIdx.x] = red[0];
}
__global__ void finsum_k(float* part) {  // -> part[65]
  __shared__ float red[64];
  red[threadIdx.x] = part[threadIdx.x]; __syncthreads();
  for (int s = 32; s > 0; s >>= 1) {
    if (threadIdx.x < s) red[threadIdx.x] += red[threadIdx.x + s];
    __syncthreads();
  }
  if (threadIdx.x == 0) part[65] = red[0];
}
__global__ void writeout_k(const float* x, int n, const float* mx, const float* sm,
                           float* out) {
  for (int i = blockIdx.x * blockDim.x + threadIdx.x; i < n; i += gridDim.x * blockDim.x)
    out[i] = x[i] - mx[0] - logf(sm[0]);
}

// ---------------------------------------------------------------------------
extern "C" void kernel_launch(void* const* d_in, const int* in_sizes, int n_in,
                              void* d_out, int out_size, void* d_ws, size_t ws_size,
                              hipStream_t stream) {
  (void)n_in; (void)out_size; (void)ws_size;
  const float* game_x  = (const float*)d_in[0];
  const float* state_x = (const float*)d_in[1];
  const float* pc_x    = (const float*)d_in[2];
  const int E_vv = in_sizes[3] / 2;
  const int* rvv = (const int*)d_in[3]; const int* cvv = rvv + E_vv;
  const int* evv = (const int*)d_in[4];
  const int E_h = in_sizes[5] / 2;
  const int* rh = (const int*)d_in[5]; const int* ch = rh + E_h;
  const float* ea_h = (const float*)d_in[6];
  const int E_i = in_sizes[7] / 2;
  const int* ri = (const int*)d_in[7]; const int* ci = ri + E_i;
  const int E_ss = in_sizes[8] / 2;
  const int* rs = (const int*)d_in[8]; const int* cs = rs + E_ss;
  const int E_pp = in_sizes[9] / 2;
  const int* rpp = (const int*)d_in[9]; const int* cpp = rpp + E_pp;
  const int E_ps = in_sizes[10] / 2;
  const int* rps = (const int*)d_in[10]; const int* cps = rps + E_ps;
  const float* tag10_w = (const float*)d_in[12];
  const float* tag10_b = (const float*)d_in[13];
  const float* rgcn_w = (const float*)d_in[14];
  const float* rgcn_root = (const float*)d_in[15];
  const float* rgcn_b = (const float*)d_in[16];
  const float* gcn1_w = (const float*)d_in[17]; const float* gcn1_b = (const float*)d_in[18];
  const float* gcn2_w = (const float*)d_in[19]; const float* gcn2_b = (const float*)d_in[20];
  const float* rg_key_w = (const float*)d_in[21]; const float* rg_key_b = (const float*)d_in[22];
  const float* rg_query_w = (const float*)d_in[23]; const float* rg_query_b = (const float*)d_in[24];
  const float* rg_value_w = (const float*)d_in[25]; const float* rg_value_b = (const float*)d_in[26];
  const float* rg_edge_w = (const float*)d_in[27];
  const float* rg_skip_w = (const float*)d_in[28]; const float* rg_b = (const float*)d_in[29];
  const float* sage32_lw = (const float*)d_in[30]; const float* sage32_lb = (const float*)d_in[31];
  const float* sage32_rw = (const float*)d_in[32];
  const float* sage4_lw = (const float*)d_in[33]; const float* sage4_lb = (const float*)d_in[34];
  const float* sage4_rw = (const float*)d_in[35];
  const float* sage42_lw = (const float*)d_in[36]; const float* sage42_lb = (const float*)d_in[37];
  const float* sage42_rw = (const float*)d_in[38];
  const float* pcs_lw = (const float*)d_in[39]; const float* pcs_lb = (const float*)d_in[40];
  const float* pcs_rw = (const float*)d_in[41];
  const float* tag2_w = (const float*)d_in[42]; const float* tag2_b = (const float*)d_in[43];
  const float* sage5_lw = (const float*)d_in[44]; const float* sage5_lb = (const float*)d_in[45];
  const float* sage5_rw = (const float*)d_in[46];
  const float* lin_w = (const float*)d_in[47]; const float* lin_b = (const float*)d_in[48];
  const float* linl_w = (const float*)d_in[49]; const float* linl_b = (const float*)d_in[50];

  // ---------------- workspace layout (floats) ----------------
  float* ws = (float*)d_ws;
  size_t o = 0;
  float* A0 = ws + o; o += (size_t)NGAME * 128;
  float* A1 = ws + o; o += (size_t)NGAME * 128;
  float* A2 = ws + o; o += (size_t)NGAME * 128;
  float* S0 = ws + o; o += (size_t)NSTATE * 128;
  float* S1 = ws + o; o += (size_t)NSTATE * 128;
  float* S2 = ws + o; o += (size_t)NSTATE * 128;
  float* S3 = ws + o; o += (size_t)NSTATE * 128;
  float* P0 = ws + o; o += (size_t)NPC * 128;
  float* P1 = ws + o; o += (size_t)NPC * 128;
  float* H7a = ws + o; o += (size_t)NGAME * 7;
  float* H7b = ws + o; o += (size_t)NGAME * 7;
  float* DEG = ws + o; o += 50000;
  float* CNT3 = ws + o; o += (size_t)NGAME * 3;
  float* SLIN = ws + o; o += (size_t)NSTATE * 32;
  float* LOG = ws + o; o += NSTATE;
  float* PART = ws + o; o += 128;

  auto nblk = [](long long total) {
    long long b = (total + 255) / 256;
    if (b > (1LL << 20)) b = 1LL << 20;
    if (b < 1) b = 1;
    return (unsigned)b;
  };
#define ZERO(p, n) zero_k<<<nblk(n), 256, 0, stream>>>(p, (long long)(n))
  const long long NG128 = (long long)NGAME * 128;
  const long long NS128 = (long long)NSTATE * 128;
  const long long NP128 = (long long)NPC * 128;

  // ================= Stage A: TAGConv(game) =================
  ZERO(DEG, NGAME);
  count_deg_k<<<nblk(E_vv), 256, 0, stream>>>(cvv, E_vv, DEG);
  dinv_k<<<nblk(NGAME), 256, 0, stream>>>(DEG, NGAME, 0.0f);
  gemm_smallK<<<nblk(NG128), 256, 0, stream>>>(game_x, tag10_w, tag10_b, A0, NGAME, 7, 0);
  {
    const float* hs = game_x; float* hd = H7a;
    for (int k = 1; k < 4; ++k) {
      ZERO(hd, (long long)NGAME * 7);
      prop_norm_k<<<nblk((long long)E_vv * 7), 256, 0, stream>>>(hs, rvv, cvv, DEG, hd,
                                                                 (long long)E_vv * 7, 7);
      gemm_smallK<<<nblk(NG128), 256, 0, stream>>>(hd, tag10_w + (size_t)k * 7 * 128,
                                                   nullptr, A0, NGAME, 7, 1);
      hs = hd; hd = (hd == H7a) ? H7b : H7a;
    }
  }
  relu_k<<<nblk(NG128), 256, 0, stream>>>(A0, NG128);  // g = A0

  // ================= Stage B: RGCNConv =================
  ZERO(CNT3, (long long)NGAME * 3);
  rgcn_count_k<<<nblk(E_vv), 256, 0, stream>>>(cvv, evv, E_vv, CNT3);
  gemm128_wmma<8, false><<<(NGAME + 127) / 128, 256, 0, stream>>>(A0, rgcn_root, rgcn_b,
                                                                  nullptr, A1, NGAME);
  for (int r = 0; r < 3; ++r) {
    gemm128_wmma<8, false><<<(NGAME + 127) / 128, 256, 0, stream>>>(
        A0, rgcn_w + (size_t)r * 128 * 128, nullptr, nullptr, A2, NGAME);
    rgcn_scatter_k<<<nblk((long long)E_vv * 128), 256, 0, stream>>>(
        A2, rvv, cvv, evv, CNT3, r, A1, (long long)E_vv * 128);
  }
  relu_k<<<nblk(NG128), 256, 0, stream>>>(A1, NG128);  // g = A1; A0,A2 free

  // ================= Stage C: two GCN layers on pc =================
  ZERO(DEG, NPC);
  count_deg_k<<<nblk(E_pp), 256, 0, stream>>>(cpp, E_pp, DEG);
  dinv_k<<<nblk(NPC), 256, 0, stream>>>(DEG, NPC, 1.0f);  // self loops
  gemm_smallK<<<nblk(NP128), 256, 0, stream>>>(pc_x, gcn1_w, nullptr, P0, NPC, 5, 0);
  ZERO(P1, NP128);
  prop_norm_k<<<nblk((long long)E_pp * 128), 256, 0, stream>>>(P0, rpp, cpp, DEG, P1,
                                                               (long long)E_pp * 128, 128);
  gcn_epi_k<<<nblk(NP128), 256, 0, stream>>>(P1, P0, DEG, gcn1_b, NP128);  // p = P1
  gemm128_wmma<8, false><<<(NPC + 127) / 128, 256, 0, stream>>>(P1, gcn2_w, nullptr,
                                                                nullptr, P0, NPC);
  ZERO(P1, NP128);
  prop_norm_k<<<nblk((long long)E_pp * 128), 256, 0, stream>>>(P0, rpp, cpp, DEG, P1,
                                                               (long long)E_pp * 128, 128);
  gcn_epi_k<<<nblk(NP128), 256, 0, stream>>>(P1, P0, DEG, gcn2_b, NP128);  // p = P1

  // ================= Stage D: ResGatedGraphConv (game->state) =================
  gemm_smallK<<<nblk(NS128), 256, 0, stream>>>(state_x, rg_key_w, rg_key_b, S2, NSTATE, 6, 0);
  gemm128_wmma<8, false><<<(NGAME + 127) / 128, 256, 0, stream>>>(A1, rg_query_w, rg_query_b,
                                                                  nullptr, A0, NGAME);  // q
  gemm128_wmma<8, false><<<(NGAME + 127) / 128, 256, 0, stream>>>(A1, rg_value_w, rg_value_b,
                                                                  nullptr, A2, NGAME);  // v
  ZERO(S0, NS128);
  gated_k<<<nblk((long long)E_h * 128), 256, 0, stream>>>(S2, A0, A2, ea_h, rg_edge_w, rh, ch,
                                                          S0, (long long)E_h * 128);
  gemm_smallK<<<nblk(NS128), 256, 0, stream>>>(state_x, rg_skip_w, rg_b, S1, NSTATE, 6, 0);
  add_relu_k<<<nblk(NS128), 256, 0, stream>>>(S0, S1, NS128);  // s = S0

  // ================= Stage E: SAGE chain =================
  // sage32 (edges rh/ch, src g)
  ZERO(DEG, NSTATE);
  count_deg_k<<<nblk(E_h), 256, 0, stream>>>(ch, E_h, DEG);
  ZERO(S1, NS128);
  sum_rows_k<<<nblk((long long)E_h * 128), 256, 0, stream>>>(A1, rh, ch, S1, (long long)E_h * 128);
  div_cnt_k<<<nblk(NS128), 256, 0, stream>>>(S1, DEG, NS128);
  gemm128_wmma<8, false><<<(NSTATE + 127) / 128, 256, 0, stream>>>(S1, sage32_lw, sage32_lb,
                                                                   nullptr, S2, NSTATE);
  gemm128_wmma<8, false><<<(NSTATE + 127) / 128, 256, 0, stream>>>(S0, sage32_rw, nullptr,
                                                                   S2, S2, NSTATE);
  l2norm_relu_k<<<NSTATE, 128, 0, stream>>>(S2, NSTATE);  // s = S2
  // sage4 (edges ri/ci, src g)
  ZERO(DEG, NSTATE);
  count_deg_k<<<nblk(E_i), 256, 0, stream>>>(ci, E_i, DEG);
  ZERO(S0, NS128);
  sum_rows_k<<<nblk((long long)E_i * 128), 256, 0, stream>>>(A1, ri, ci, S0, (long long)E_i * 128);
  div_cnt_k<<<nblk(NS128), 256, 0, stream>>>(S0, DEG, NS128);
  gemm128_wmma<8, false><<<(NSTATE + 127) / 128, 256, 0, stream>>>(S0, sage4_lw, sage4_lb,
                                                                   nullptr, S1, NSTATE);
  gemm128_wmma<8, false><<<(NSTATE + 127) / 128, 256, 0, stream>>>(S2, sage4_rw, nullptr,
                                                                   S1, S1, NSTATE);
  l2norm_relu_k<<<NSTATE, 128, 0, stream>>>(S1, NSTATE);  // s = S1
  // sage42 (same edges/deg, src g)
  ZERO(S0, NS128);
  sum_rows_k<<<nblk((long long)E_i * 128), 256, 0, stream>>>(A1, ri, ci, S0, (long long)E_i * 128);
  div_cnt_k<<<nblk(NS128), 256, 0, stream>>>(S0, DEG, NS128);
  gemm128_wmma<8, false><<<(NSTATE + 127) / 128, 256, 0, stream>>>(S0, sage42_lw, sage42_lb,
                                                                   nullptr, S3, NSTATE);
  gemm128_wmma<8, false><<<(NSTATE + 127) / 128, 256, 0, stream>>>(S1, sage42_rw, nullptr,
                                                                   S3, S3, NSTATE);
  l2norm_relu_k<<<NSTATE, 128, 0, stream>>>(S3, NSTATE);  // s = S3
  // pcs (edges rps/cps, src p)
  ZERO(DEG, NSTATE);
  count_deg_k<<<nblk(E_ps), 256, 0, stream>>>(cps, E_ps, DEG);
  ZERO(S0, NS128);
  sum_rows_k<<<nblk((long long)E_ps * 128), 256, 0, stream>>>(P1, rps, cps, S0,
                                                              (long long)E_ps * 128);
  div_cnt_k<<<nblk(NS128), 256, 0, stream>>>(S0, DEG, NS128);
  gemm128_wmma<8, false><<<(NSTATE + 127) / 128, 256, 0, stream>>>(S0, pcs_lw, pcs_lb,
                                                                   nullptr, S1, NSTATE);
  gemm128_wmma<8, false><<<(NSTATE + 127) / 128, 256, 0, stream>>>(S3, pcs_rw, nullptr,
                                                                   S1, S1, NSTATE);
  l2norm_relu_k<<<NSTATE, 128, 0, stream>>>(S1, NSTATE);  // s = S1

  // ================= Stage F: TAGConv(state) =================
  ZERO(DEG, NSTATE);
  count_deg_k<<<nblk(E_ss), 256, 0, stream>>>(cs, E_ss, DEG);
  dinv_k<<<nblk(NSTATE), 256, 0, stream>>>(DEG, NSTATE, 0.0f);
  gemm128_wmma<8, false><<<(NSTATE + 127) / 128, 256, 0, stream>>>(S1, tag2_w, tag2_b,
                                                                   nullptr, S0, NSTATE);
  ZERO(S2, NS128);
  prop_norm_k<<<nblk((long long)E_ss * 128), 256, 0, stream>>>(S1, rs, cs, DEG, S2,
                                                               (long long)E_ss * 128, 128);
  gemm128_wmma<8, false><<<(NSTATE + 127) / 128, 256, 0, stream>>>(
      S2, tag2_w + (size_t)1 * 128 * 128, nullptr, S0, S0, NSTATE);
  ZERO(S3, NS128);
  prop_norm_k<<<nblk((long long)E_ss * 128), 256, 0, stream>>>(S2, rs, cs, DEG, S3,
                                                               (long long)E_ss * 128, 128);
  gemm128_wmma<8, false><<<(NSTATE + 127) / 128, 256, 0, stream>>>(
      S3, tag2_w + (size_t)2 * 128 * 128, nullptr, S0, S0, NSTATE);
  relu_k<<<nblk(NS128), 256, 0, stream>>>(S0, NS128);  // s = S0

  // ================= Stage G: sage5 (edges rs/cs, src s) =================
  ZERO(DEG, NSTATE);
  count_deg_k<<<nblk(E_ss), 256, 0, stream>>>(cs, E_ss, DEG);
  ZERO(S2, NS128);
  sum_rows_k<<<nblk((long long)E_ss * 128), 256, 0, stream>>>(S0, rs, cs, S2,
                                                              (long long)E_ss * 128);
  div_cnt_k<<<nblk(NS128), 256, 0, stream>>>(S2, DEG, NS128);
  gemm128_wmma<8, false><<<(NSTATE + 127) / 128, 256, 0, stream>>>(S2, sage5_lw, sage5_lb,
                                                                   nullptr, S1, NSTATE);
  gemm128_wmma<8, false><<<(NSTATE + 127) / 128, 256, 0, stream>>>(S0, sage5_rw, nullptr,
                                                                   S1, S1, NSTATE);
  l2norm_relu_k<<<NSTATE, 128, 0, stream>>>(S1, NSTATE);  // s = S1

  // ================= Stage H/I/J: lin -> linl -> log_softmax =================
  gemm128_wmma<2, true><<<(NSTATE + 127) / 128, 256, 0, stream>>>(S1, lin_w, lin_b,
                                                                  nullptr, SLIN, NSTATE);
  rowdot32_k<<<(NSTATE + 255) / 256, 256, 0, stream>>>(SLIN, linl_w, linl_b, LOG, NSTATE);
  redmax_k<<<64, 256, 0, stream>>>(LOG, NSTATE, PART);
  finmax_k<<<1, 64, 0, stream>>>(PART);
  redsum_k<<<64, 256, 0, stream>>>(LOG, NSTATE, PART + 64, PART);
  finsum_k<<<1, 64, 0, stream>>>(PART);
  writeout_k<<<(NSTATE + 255) / 256, 256, 0, stream>>>(LOG, NSTATE, PART + 64, PART + 65,
                                                       (float*)d_out);
#undef ZERO
}
